// RSSM_61959198212676
// MI455X (gfx1250) — compile-verified, matrix operand order checked by reading, not run on hardware
//
#include <hip/hip_runtime.h>

// ---------------------------------------------------------------------------
// RSSM rollout for MI455X (gfx1250), wave32, WMMA bf16 16x16x32, fp32 accum.
//
// Packed bf16 weights in d_ws (dense K stride = KPAD = 1568):
//   Wg   [1536][1568] : [W_hh | W_ih(:,:1024) | W_ih(:,1024:1026) | 0-pad]
//   Wp1  [ 512][1568] : [post_W1 (deter|embed|goal) | 0-pad]
//   Wpr1 [ 512][ 512] : prior_W1
//   Wpr2 [1024][ 512] : prior_W2
//   Wq2  [1024][ 512] : post_W2
// LDS A buffer sIn[16][KROW] (KROW=1576, bank-conflict-free row stride),
// logical cols = [h(512) | stoch-or-embed(1024) | act-or-goal(2) | 0-pad].
// Recurrent h kept fp32 in LDS (sH); bf16 copy regenerated per step.
// ---------------------------------------------------------------------------

typedef __attribute__((ext_vector_type(16))) __bf16 v16bf;
typedef __attribute__((ext_vector_type(8)))  float  v8f;

#define DETER 512
#define EMBED 1024
#define STOCHN 1024
#define ACTN 2
#define GOALN 2
#define BSZ 256
#define TN 64
#define KPAD 1568            // dense logical K (49 k-tiles of 32)
#define KROW 1576            // LDS row stride: 788 dwords = 20 mod 64 banks
#define HROW 520             // hidden LDS row stride: 260 dwords = 4 mod 64

// ws section sizes (bf16 elements)
#define SZ_WG   (1536*KPAD)
#define SZ_WP1  (512*KPAD)
#define SZ_WPR1 (512*512)
#define SZ_WPR2 (1024*512)
#define SZ_WQ2  (1024*512)
#define SZ_TOTAL (SZ_WG + SZ_WP1 + SZ_WPR1 + SZ_WPR2 + SZ_WQ2)

// ---- fragment load: elements 0..7 <- p[0..7], 8..15 <- p[16..23] (16B each)
__device__ __forceinline__ v16bf frag_ld(const __bf16* p) {
    union { v16bf v; uint4 q[2]; } f;
    f.q[0] = *reinterpret_cast<const uint4*>(p);
    f.q[1] = *reinterpret_cast<const uint4*>(p + 16);
    return f.v;
}

__device__ __forceinline__ v8f wmma_bf16(v16bf a, v16bf b, v8f c) {
    return __builtin_amdgcn_wmma_f32_16x16x32_bf16(
        false, a, false, b, (short)0, c, false, false);
}

__device__ __forceinline__ unsigned hashu(unsigned a, unsigned b,
                                          unsigned c, unsigned d) {
    unsigned x = a * 0x9E3779B9u ^ b;
    x ^= x >> 16; x *= 0x85EBCA6Bu;
    x ^= c * 0xC2B2AE35u;
    x ^= x >> 13; x *= 0x27D4EB2Fu;
    x ^= d * 0x165667B1u;
    x ^= x >> 16;
    return x;
}

// ---------------------------------------------------------------------------
// One-time fp32 -> bf16 weight repack into d_ws.
// ---------------------------------------------------------------------------
__global__ void rssm_prep_weights(const float* __restrict__ W_ih,
                                  const float* __restrict__ W_hh,
                                  const float* __restrict__ post_W1,
                                  const float* __restrict__ prior_W1,
                                  const float* __restrict__ prior_W2,
                                  const float* __restrict__ post_W2,
                                  __bf16* __restrict__ ws) {
    size_t i = (size_t)blockIdx.x * blockDim.x + threadIdx.x;
    if (i >= (size_t)SZ_TOTAL) return;
    float v = 0.0f;
    if (i < SZ_WG) {
        int row = (int)(i / KPAD), col = (int)(i % KPAD);
        if (col < 512)       v = W_hh[row * 512 + col];
        else if (col < 1536) v = W_ih[row * 1026 + (col - 512)];
        else if (col < 1538) v = W_ih[row * 1026 + 1024 + (col - 1536)];
    } else if (i < SZ_WG + SZ_WP1) {
        size_t j = i - SZ_WG;
        int row = (int)(j / KPAD), col = (int)(j % KPAD);
        if (col < 1538) v = post_W1[row * 1538 + col];
    } else if (i < SZ_WG + SZ_WP1 + SZ_WPR1) {
        v = prior_W1[i - (SZ_WG + SZ_WP1)];
    } else if (i < SZ_WG + SZ_WP1 + SZ_WPR1 + SZ_WPR2) {
        v = prior_W2[i - (SZ_WG + SZ_WP1 + SZ_WPR1)];
    } else {
        v = post_W2[i - (SZ_WG + SZ_WP1 + SZ_WPR1 + SZ_WPR2)];
    }
    ws[i] = (__bf16)v;
}

// ---------------------------------------------------------------------------
// Persistent rollout: each workgroup owns 16 batch rows for all 64 steps.
// 256 threads = 8 waves; wave w owns column tiles {w, w+8, (w+16, w+24)}.
// ---------------------------------------------------------------------------
__global__ __launch_bounds__(256, 1)
void rssm_rollout(const float* __restrict__ embeds,
                  const float* __restrict__ actions,
                  const float* __restrict__ goals,
                  const float* __restrict__ b_ih,
                  const float* __restrict__ b_hh,
                  const float* __restrict__ prior_b1,
                  const float* __restrict__ prior_b2,
                  const float* __restrict__ post_b1,
                  const float* __restrict__ post_b2,
                  const unsigned* __restrict__ keys,
                  const __bf16* __restrict__ ws,
                  float* __restrict__ out_d,
                  float* __restrict__ out_s,
                  float* __restrict__ out_pl,
                  float* __restrict__ out_prl) {
    __shared__ __align__(16) __bf16 sIn[16 * KROW];      // ~50 KB  A operand
    __shared__ __align__(16) __bf16 sHidA[16 * HROW];    // ~17 KB  prior hidden
    __shared__ __align__(16) __bf16 sHidB[16 * HROW];    // ~17 KB  post hidden
    __shared__ __align__(16) float  sLogit[16 * STOCHN]; //  64 KB  post logits
    __shared__ __align__(16) float  sH[16 * DETER];      //  32 KB  fp32 deter

    const int tid  = threadIdx.x;
    const int wv   = tid >> 5;        // wave id 0..7
    const int lane = tid & 31;
    const int lrow = lane & 15;
    const int half = lane >> 4;
    const int b0   = blockIdx.x * 16;

    const __bf16* Wg   = ws;
    const __bf16* Wp1  = Wg + SZ_WG;
    const __bf16* Wpr1 = Wp1 + SZ_WP1;
    const __bf16* Wpr2 = Wpr1 + SZ_WPR1;
    const __bf16* Wq2  = Wpr2 + SZ_WPR2;

    // ---- init: zero A-buffer (permanent K padding) and fp32 h ----
    for (int i = tid; i < 16 * KROW; i += 256) sIn[i] = (__bf16)0.0f;
    for (int i = tid; i < 16 * DETER; i += 256) sH[i] = 0.0f;
    __syncthreads();
    // stoch0: uniform probs -> straight-through value is exactly one-hot
    for (int g = tid; g < 512; g += 256) {
        int m = g >> 5, c = g & 31;
        unsigned h = hashu(keys[0], keys[1], (unsigned)(b0 + m), (unsigned)c);
        sIn[m * KROW + 512 + c * 32 + (int)(h & 31)] = (__bf16)1.0f;
    }
    __syncthreads();

    for (int t = 0; t < TN; ++t) {
        // ---- stage actions for this step (stoch/h already in sIn) ----
        if (tid < 32) {
            int m = tid >> 1, a = tid & 1;
            sIn[m * KROW + 1536 + a] =
                (__bf16)actions[((size_t)(b0 + m) * TN + t) * ACTN + a];
        }
        __syncthreads();

        // ===== fused GRU: one A-load feeds r,z,n-gate WMMAs ==============
        // Ping-pong double buffer, pairs of k-tiles, no register rotation.
#pragma unroll 1
        for (int i = 0; i < 4; ++i) {
            const int n = (wv + 8 * i) * 16 + lrow;       // deter column
            const __bf16* pa  = sIn + lrow * KROW + half * 8;
            const __bf16* pbR = Wg + (size_t)n * KPAD + half * 8;
            const __bf16* pbZ = pbR + (size_t)512 * KPAD;
            const __bf16* pbN = pbR + (size_t)1024 * KPAD;

            v8f aR = {}, aZ = {}, aI = {}, aH = {};
            v16bf a_0, bR_0, bZ_0, bN_0, a_1, bR_1, bZ_1, bN_1;
            a_0  = frag_ld(pa);   bR_0 = frag_ld(pbR);
            bZ_0 = frag_ld(pbZ);  bN_0 = frag_ld(pbN);

            // k-tiles 0..15 : n-gate accumulates h-path (aH)
#pragma unroll 1
            for (int d = 0; d < 8; ++d) {
                const int o1 = (2 * d + 1) * 32, o2 = (2 * d + 2) * 32;
                __builtin_prefetch(pbR + o2 + 512, 0, 3);
                __builtin_prefetch(pbZ + o2 + 512, 0, 3);
                __builtin_prefetch(pbN + o2 + 512, 0, 3);
                a_1  = frag_ld(pa + o1);   bR_1 = frag_ld(pbR + o1);
                bZ_1 = frag_ld(pbZ + o1);  bN_1 = frag_ld(pbN + o1);
                aR = wmma_bf16(a_0, bR_0, aR);
                aZ = wmma_bf16(a_0, bZ_0, aZ);
                aH = wmma_bf16(a_0, bN_0, aH);
                a_0  = frag_ld(pa + o2);   bR_0 = frag_ld(pbR + o2);
                bZ_0 = frag_ld(pbZ + o2);  bN_0 = frag_ld(pbN + o2);
                aR = wmma_bf16(a_1, bR_1, aR);
                aZ = wmma_bf16(a_1, bZ_1, aZ);
                aH = wmma_bf16(a_1, bN_1, aH);
            }
            // k-tiles 16..47 : n-gate accumulates x-path (aI)
#pragma unroll 1
            for (int d = 8; d < 24; ++d) {
                const int o1 = (2 * d + 1) * 32, o2 = (2 * d + 2) * 32;
                __builtin_prefetch(pbR + o2 + 512, 0, 3);
                __builtin_prefetch(pbZ + o2 + 512, 0, 3);
                __builtin_prefetch(pbN + o2 + 512, 0, 3);
                a_1  = frag_ld(pa + o1);   bR_1 = frag_ld(pbR + o1);
                bZ_1 = frag_ld(pbZ + o1);  bN_1 = frag_ld(pbN + o1);
                aR = wmma_bf16(a_0, bR_0, aR);
                aZ = wmma_bf16(a_0, bZ_0, aZ);
                aI = wmma_bf16(a_0, bN_0, aI);
                a_0  = frag_ld(pa + o2);   bR_0 = frag_ld(pbR + o2);
                bZ_0 = frag_ld(pbZ + o2);  bN_0 = frag_ld(pbN + o2);
                aR = wmma_bf16(a_1, bR_1, aR);
                aZ = wmma_bf16(a_1, bZ_1, aZ);
                aI = wmma_bf16(a_1, bN_1, aI);
            }
            // peeled k-tile 48
            aR = wmma_bf16(a_0, bR_0, aR);
            aZ = wmma_bf16(a_0, bZ_0, aZ);
            aI = wmma_bf16(a_0, bN_0, aI);

            const float br = b_ih[n] + b_hh[n];
            const float bz = b_ih[512 + n] + b_hh[512 + n];
            const float bi = b_ih[1024 + n];
            const float bh = b_hh[1024 + n];
#pragma unroll
            for (int e = 0; e < 8; ++e) {
                const int m = e + 8 * half;
                const float hold = sH[m * DETER + n];   // owner-wave column
                const float r  = 1.0f / (1.0f + __expf(-(aR[e] + br)));
                const float z  = 1.0f / (1.0f + __expf(-(aZ[e] + bz)));
                const float nn = tanhf(aI[e] + bi + r * (aH[e] + bh));
                sH[m * DETER + n] = (1.0f - z) * nn + z * hold;
            }
        }
        __syncthreads();   // all sIn reads + sH writes complete

        // ---- disjoint sIn refresh: h(bf16)+out_d | embed | goal ----
        for (int i = tid; i < 16 * DETER; i += 256) {
            const int m = i >> 9, col = i & 511;
            const float hv = sH[i];
            sIn[m * KROW + col] = (__bf16)hv;
            out_d[((size_t)(b0 + m) * TN + t) * DETER + col] = hv;  // coalesced
        }
        for (int i = tid; i < 16 * 1024; i += 256) {
            const int m = i >> 10, col = i & 1023;
            sIn[m * KROW + 512 + col] =
                (__bf16)embeds[((size_t)(b0 + m) * TN + t) * EMBED + col];
        }
        if (tid < 32) {
            int m = tid >> 1, gcol = tid & 1;
            sIn[m * KROW + 1536 + gcol] =
                (__bf16)goals[((size_t)(b0 + m) * TN + t) * GOALN + gcol];
        }
        __syncthreads();

        // ===== layer 1: post (full K) fused with prior (K = h range) =====
#pragma unroll 1
        for (int i = 0; i < 4; ++i) {
            const int n = (wv + 8 * i) * 16 + lrow;
            const __bf16* pa  = sIn + lrow * KROW + half * 8;
            const __bf16* pbQ = Wp1 + (size_t)n * KPAD + half * 8;
            const __bf16* pbP = Wpr1 + (size_t)n * 512 + half * 8;

            v8f ap = {}, aq = {};
            v16bf a_0, bq_0, bp_0, a_1, bq_1, bp_1;
            a_0 = frag_ld(pa); bq_0 = frag_ld(pbQ); bp_0 = frag_ld(pbP);

            // k-tiles 0..15: both prior and post
#pragma unroll 1
            for (int d = 0; d < 8; ++d) {
                const int o1 = (2 * d + 1) * 32, o2 = (2 * d + 2) * 32;
                __builtin_prefetch(pbQ + o2 + 512, 0, 3);
                a_1 = frag_ld(pa + o1); bq_1 = frag_ld(pbQ + o1);
                bp_1 = frag_ld(pbP + o1);
                aq = wmma_bf16(a_0, bq_0, aq);
                ap = wmma_bf16(a_0, bp_0, ap);
                a_0 = frag_ld(pa + o2); bq_0 = frag_ld(pbQ + o2);
                bp_0 = frag_ld(pbP + o2);   // d=7 loads dummy (in-bounds, unused)
                aq = wmma_bf16(a_1, bq_1, aq);
                ap = wmma_bf16(a_1, bp_1, ap);
            }
            // k-tiles 16..47: post only
#pragma unroll 1
            for (int d = 8; d < 24; ++d) {
                const int o1 = (2 * d + 1) * 32, o2 = (2 * d + 2) * 32;
                __builtin_prefetch(pbQ + o2 + 512, 0, 3);
                a_1 = frag_ld(pa + o1); bq_1 = frag_ld(pbQ + o1);
                aq = wmma_bf16(a_0, bq_0, aq);
                a_0 = frag_ld(pa + o2); bq_0 = frag_ld(pbQ + o2);
                aq = wmma_bf16(a_1, bq_1, aq);
            }
            aq = wmma_bf16(a_0, bq_0, aq);   // peeled k-tile 48

            const float bp = prior_b1[n], bq = post_b1[n];
#pragma unroll
            for (int e = 0; e < 8; ++e) {
                const int m = e + 8 * half;
                const float x = ap[e] + bp;
                sHidA[m * HROW + n] = (__bf16)(x > 0.0f ? x : __expf(x) - 1.0f);
                const float y = aq[e] + bq;
                sHidB[m * HROW + n] = (__bf16)(y > 0.0f ? y : __expf(y) - 1.0f);
            }
        }
        __syncthreads();

        // ===== layer 2: prior/post logits (N = 1024 each), 16 k-tiles =====
#pragma unroll 1
        for (int i = 0; i < 8; ++i) {
            const int n = (wv + 8 * i) * 16 + lrow;
            const __bf16* paA = sHidA + lrow * HROW + half * 8;
            const __bf16* paB = sHidB + lrow * HROW + half * 8;
            const __bf16* pbP = Wpr2 + (size_t)n * 512 + half * 8;
            const __bf16* pbQ = Wq2 + (size_t)n * 512 + half * 8;

            v8f apr = {}, apo = {};
            v16bf aA_0, aB_0, bP_0, bQ_0, aA_1, aB_1, bP_1, bQ_1;
            aA_0 = frag_ld(paA); aB_0 = frag_ld(paB);
            bP_0 = frag_ld(pbP); bQ_0 = frag_ld(pbQ);
#pragma unroll 1
            for (int d = 0; d < 7; ++d) {        // wmma k-tiles 0..13
                const int o1 = (2 * d + 1) * 32, o2 = (2 * d + 2) * 32;
                aA_1 = frag_ld(paA + o1); aB_1 = frag_ld(paB + o1);
                bP_1 = frag_ld(pbP + o1); bQ_1 = frag_ld(pbQ + o1);
                apr = wmma_bf16(aA_0, bP_0, apr);
                apo = wmma_bf16(aB_0, bQ_0, apo);
                aA_0 = frag_ld(paA + o2); aB_0 = frag_ld(paB + o2);
                bP_0 = frag_ld(pbP + o2); bQ_0 = frag_ld(pbQ + o2);
                apr = wmma_bf16(aA_1, bP_1, apr);
                apo = wmma_bf16(aB_1, bQ_1, apo);
            }
            aA_1 = frag_ld(paA + 15 * 32); aB_1 = frag_ld(paB + 15 * 32);
            bP_1 = frag_ld(pbP + 15 * 32); bQ_1 = frag_ld(pbQ + 15 * 32);
            apr = wmma_bf16(aA_0, bP_0, apr);    // k-tile 14
            apo = wmma_bf16(aB_0, bQ_0, apo);
            apr = wmma_bf16(aA_1, bP_1, apr);    // k-tile 15
            apo = wmma_bf16(aB_1, bQ_1, apo);

            const float bpr = prior_b2[n], bpo = post_b2[n];
#pragma unroll
            for (int e = 0; e < 8; ++e) {
                const int m = e + 8 * half;
                const float pr = apr[e] + bpr;
                const float po = apo[e] + bpo;
                out_prl[((size_t)(b0 + m) * TN + t) * STOCHN + n] = pr;
                out_pl [((size_t)(b0 + m) * TN + t) * STOCHN + n] = po;
                sLogit[m * STOCHN + n] = po;
            }
        }
        __syncthreads();

        // ---- softmax + unimix + categorical + straight-through ----
        for (int g = tid; g < 512; g += 256) {
            const int m = g >> 5, c = g & 31;
            const float* lg = &sLogit[m * STOCHN + c * 32];
            float mx = -1e30f;
#pragma unroll
            for (int k = 0; k < 32; ++k) mx = fmaxf(mx, lg[k]);
            float p[32], s = 0.0f;
#pragma unroll
            for (int k = 0; k < 32; ++k) { p[k] = __expf(lg[k] - mx); s += p[k]; }
            const float inv = 1.0f / s;
            float psum = 0.0f;
#pragma unroll
            for (int k = 0; k < 32; ++k) {
                p[k] = 0.99f * (p[k] * inv) + 0.01f / 32.0f;
                psum += p[k];
            }
            const float inv2 = 1.0f / (psum + 1e-8f);
            const unsigned h = hashu(keys[2 * (t + 1)], keys[2 * (t + 1) + 1],
                                     (unsigned)(b0 + m), (unsigned)c);
            const float u = (float)(h >> 8) * (1.0f / 16777216.0f);
            float cdf = 0.0f; int idx = 0;
#pragma unroll
            for (int k = 0; k < 31; ++k) {
                cdf += p[k] * inv2;
                idx += (u >= cdf) ? 1 : 0;
            }
#pragma unroll
            for (int k = 0; k < 32; ++k) {
                const float pk = p[k] * inv2;
                const float val = (((k == idx) ? 1.0f : 0.0f) + pk) - pk;
                out_s[((size_t)(b0 + m) * TN + t) * STOCHN + c * 32 + k] = val;
                sIn[m * KROW + 512 + c * 32 + k] = (__bf16)val;  // next-step stoch
            }
        }
        __syncthreads();
    }
}

// ---------------------------------------------------------------------------
extern "C" void kernel_launch(void* const* d_in, const int* in_sizes, int n_in,
                              void* d_out, int out_size, void* d_ws, size_t ws_size,
                              hipStream_t stream) {
    const float* embeds   = (const float*)d_in[0];
    const float* actions  = (const float*)d_in[1];
    const float* goals    = (const float*)d_in[2];
    const float* W_ih     = (const float*)d_in[3];
    const float* W_hh     = (const float*)d_in[4];
    const float* b_ih     = (const float*)d_in[5];
    const float* b_hh     = (const float*)d_in[6];
    const float* prior_W1 = (const float*)d_in[7];
    const float* prior_b1 = (const float*)d_in[8];
    const float* prior_W2 = (const float*)d_in[9];
    const float* prior_b2 = (const float*)d_in[10];
    const float* post_W1  = (const float*)d_in[11];
    const float* post_b1  = (const float*)d_in[12];
    const float* post_W2  = (const float*)d_in[13];
    const float* post_b2  = (const float*)d_in[14];
    const unsigned* keys  = (const unsigned*)d_in[15];

    __bf16* wsb = (__bf16*)d_ws;

    int prep_blocks = (SZ_TOTAL + 255) / 256;
    rssm_prep_weights<<<prep_blocks, 256, 0, stream>>>(
        W_ih, W_hh, post_W1, prior_W1, prior_W2, post_W2, wsb);

    float* out_d   = (float*)d_out;
    float* out_s   = out_d + (size_t)BSZ * TN * DETER;
    float* out_pl  = out_s + (size_t)BSZ * TN * STOCHN;
    float* out_prl = out_pl + (size_t)BSZ * TN * STOCHN;

    rssm_rollout<<<BSZ / 16, 256, 0, stream>>>(
        embeds, actions, goals, b_ih, b_hh,
        prior_b1, prior_b2, post_b1, post_b2,
        keys, wsb, out_d, out_s, out_pl, out_prl);
}
